// SimpleSelfAttention_2491081032135
// MI455X (gfx1250) — matrix operand order repaired and checked by reading, not compiled, not run on hardware
//
#include <hip/hip_runtime.h>

typedef __bf16 bf16_t;
typedef __attribute__((ext_vector_type(16))) __bf16 v16bf;
typedef __attribute__((ext_vector_type(8)))  __bf16 v8bf;
typedef __attribute__((ext_vector_type(4)))  __bf16 v4bf;
typedef __attribute__((ext_vector_type(8)))  float  v8f;
typedef __attribute__((ext_vector_type(4)))  float  v4f;
typedef __attribute__((ext_vector_type(4)))  int    v4i;

#define BATCH 16
#define CDIM  1024
#define NDIM  4096

#define BM 128
#define BN 128
#define BK 32

#define GLOBAL_AS __attribute__((address_space(1)))
#define LDS_AS    __attribute__((address_space(3)))

#if __has_builtin(__builtin_amdgcn_global_load_async_to_lds_b128)
#define HAVE_ASYNC 1
#endif

// ---- async 16B global -> LDS copy (falls back to VGPR bounce if builtin absent) ----
__device__ __forceinline__ void async_cp16(const bf16_t* g, bf16_t* l) {
#ifdef HAVE_ASYNC
  __builtin_amdgcn_global_load_async_to_lds_b128(
      (GLOBAL_AS v4i*)const_cast<bf16_t*>(g),
      (LDS_AS v4i*)l, 0, 0);
#else
  *(v8bf*)l = *(const v8bf*)g;
#endif
}

__device__ __forceinline__ void wait_async0() {
#ifdef HAVE_ASYNC
#if __has_builtin(__builtin_amdgcn_s_wait_asynccnt)
  __builtin_amdgcn_s_wait_asynccnt(0);
#else
  asm volatile("s_wait_asynccnt 0x0" ::: "memory");
#endif
#endif
}

// ---------------- fp32 -> bf16 conversion (vectorized, grid-stride) ----------------
__global__ __launch_bounds__(256) void cvt_f32_bf16(const float* __restrict__ in,
                                                    bf16_t* __restrict__ out, long n4) {
  const v4f* in4 = (const v4f*)in;
  v4bf* out4 = (v4bf*)out;
  long i = (long)blockIdx.x * blockDim.x + threadIdx.x;
  long stride = (long)gridDim.x * blockDim.x;
  for (; i < n4; i += stride) {
    v4f a = in4[i];
    v4bf o;
    o[0] = (bf16_t)a[0]; o[1] = (bf16_t)a[1];
    o[2] = (bf16_t)a[2]; o[3] = (bf16_t)a[3];
    out4[i] = o;
  }
}

// ---- fused convert: read x[b] (rows x cols fp32) once, write xh (same layout, bf16)
// ---- and xhT (cols x rows, bf16) via 32x32 LDS-tiled transpose.
__global__ __launch_bounds__(256)
void cvt_and_transpose(const float* __restrict__ x, bf16_t* __restrict__ xh,
                       bf16_t* __restrict__ xt, int rows, int cols) {
  __shared__ float tile[32][33];
  const int b = blockIdx.z;
  const float* xb = x + (long)b * rows * cols;
  bf16_t* xhb = xh + (long)b * rows * cols;
  bf16_t* xtb = xt + (long)b * rows * cols;
  const int r0 = blockIdx.y * 32;   // row block (source rows)
  const int c0 = blockIdx.x * 32;   // col block (source cols)
  const int tx = threadIdx.x & 31;
  const int ty = threadIdx.x >> 5;  // 0..7
  #pragma unroll
  for (int i = 0; i < 32; i += 8) {
    float v = xb[(long)(r0 + ty + i) * cols + c0 + tx];
    tile[ty + i][tx] = v;
    xhb[(long)(r0 + ty + i) * cols + c0 + tx] = (bf16_t)v;
  }
  __syncthreads();
  #pragma unroll
  for (int i = 0; i < 32; i += 8)
    xtb[(long)(c0 + ty + i) * rows + r0 + tx] = (bf16_t)tile[tx][ty + i];
}

// Build a 16-elem bf16 fragment from two contiguous 8-elem LDS chunks.
__device__ __forceinline__ v16bf ld_frag(const bf16_t* p, int off2) {
  union U { v16bf v; v8bf h[2]; } u;
  u.h[0] = *(const v8bf*)p;
  u.h[1] = *(const v8bf*)(p + off2);
  return u.v;
}

// ---------------- NT-only tiled bf16 WMMA GEMM, double-buffered async staging -----
// C(MxN) = A(MxK, row-major lda) * Bt^T where Bt is N x K row-major (ldb).
// FINAL=false: store bf16 C (ldc). FINAL=true: Cf = gamma*acc + resid (fp32).
template <bool FINAL>
__global__ __launch_bounds__(256)
void gemm_nt_bf16_wmma(const bf16_t* __restrict__ A, long strideAb,
                       const bf16_t* __restrict__ Bt, long strideBb,
                       bf16_t* __restrict__ Cb, float* __restrict__ Cf,
                       const float* __restrict__ resid, long strideCb,
                       const float* __restrict__ gammap,
                       int K, int lda, int ldb, int ldc) {
  __shared__ bf16_t Alds[2][BM * BK];   // [buf][m][k], 8 KB each
  __shared__ bf16_t Blds[2][BN * BK];   // [buf][n][k], 8 KB each

  const int b = blockIdx.z;
  const bf16_t* Ab = A + (long)b * strideAb;
  const bf16_t* Bb = Bt + (long)b * strideBb;

  const int m0 = blockIdx.y * BM;
  const int n0 = blockIdx.x * BN;

  const int tid  = threadIdx.x;       // 0..255 (8 wave32)
  const int lane = tid & 31;
  const int wave = tid >> 5;
  const int wm = wave >> 2;           // 0..1 -> 64-row slab
  const int wn = wave & 3;            // 0..3 -> 32-col slab
  const int r    = lane & 15;
  const int half = lane >> 4;

  // staging map: 256 threads x 2 passes x 16B = 8 KB tile
  const int srow = tid >> 2;          // 0..63
  const int sc8  = (tid & 3) * 8;     // 0,8,16,24

  v8f acc[4][2] = {};

  const int nk = K / BK;

  // prologue: stage tile 0 into buffer 0
  #pragma unroll
  for (int p = 0; p < 2; ++p) {
    async_cp16(Ab + (long)(m0 + p * 64 + srow) * lda + sc8,
               &Alds[0][(p * 64 + srow) * BK + sc8]);
    async_cp16(Bb + (long)(n0 + p * 64 + srow) * ldb + sc8,
               &Blds[0][(p * 64 + srow) * BK + sc8]);
  }

  for (int kk = 0; kk < nk; ++kk) {
    const int buf = kk & 1;
    wait_async0();          // my async fills of buf are complete
    __syncthreads();        // everyone's fills complete; everyone done reading buf^1

    // prefetch next K-tile into the other buffer (overlaps with WMMAs below)
    if (kk + 1 < nk) {
      const int k0n = (kk + 1) * BK;
      const int nbuf = buf ^ 1;
      #pragma unroll
      for (int p = 0; p < 2; ++p) {
        async_cp16(Ab + (long)(m0 + p * 64 + srow) * lda + k0n + sc8,
                   &Alds[nbuf][(p * 64 + srow) * BK + sc8]);
        async_cp16(Bb + (long)(n0 + p * 64 + srow) * ldb + k0n + sc8,
                   &Blds[nbuf][(p * 64 + srow) * BK + sc8]);
      }
    }

    // ---- fragments + 8 WMMAs on current buffer ----
    v16bf afr[4], bfr[2];
    #pragma unroll
    for (int mi = 0; mi < 4; ++mi) {
      // A 16x32 bf16 layout: lane(r,half): e=0..7 -> K=half*8+e ; e=8..15 -> K=16+half*8+(e-8)
      const bf16_t* p = &Alds[buf][(wm * 64 + mi * 16 + r) * BK + half * 8];
      afr[mi] = ld_frag(p, 16);
    }
    #pragma unroll
    for (int ni = 0; ni < 2; ++ni) {
      // B 32x16 bf16 layout: lane(c,half): e=0..7 -> K=half*8+e ; e=8..15 -> K=16+half*8+(e-8), N=c (row-contiguous in LDS)
      const bf16_t* p = &Blds[buf][(wn * 32 + ni * 16 + r) * BK + half * 8];
      bfr[ni] = ld_frag(p, 16);
    }
    #pragma unroll
    for (int mi = 0; mi < 4; ++mi)
      #pragma unroll
      for (int ni = 0; ni < 2; ++ni)
        acc[mi][ni] = __builtin_amdgcn_wmma_f32_16x16x32_bf16(
            false, afr[mi], false, bfr[ni], (short)0, acc[mi][ni], false, false);
  }

  // ---- writeback. C/D layout: VGPR v -> row = half*8+v, col = lane&15 ----
  const float g = FINAL ? gammap[0] : 0.0f;
  const float* Rb = FINAL ? (resid + (long)b * strideCb) : nullptr;
  float* Ob = FINAL ? (Cf + (long)b * strideCb) : nullptr;
  bf16_t* Hb = FINAL ? nullptr : (Cb + (long)b * strideCb);

  #pragma unroll
  for (int mi = 0; mi < 4; ++mi) {
    #pragma unroll
    for (int ni = 0; ni < 2; ++ni) {
      const int col = n0 + wn * 32 + ni * 16 + r;
      #pragma unroll
      for (int v = 0; v < 8; ++v) {
        const int row = m0 + wm * 64 + mi * 16 + half * 8 + v;
        const long idx = (long)row * ldc + col;
        if (FINAL) {
          Ob[idx] = g * acc[mi][ni][v] + Rb[idx];
        } else {
          Hb[idx] = (bf16_t)acc[mi][ni][v];
        }
      }
    }
  }
}

extern "C" void kernel_launch(void* const* d_in, const int* in_sizes, int n_in,
                              void* d_out, int out_size, void* d_ws, size_t ws_size,
                              hipStream_t stream) {
  const float* x     = (const float*)d_in[0];   // [16,1024,64,64]
  const float* W     = (const float*)d_in[1];   // [1024,1024]
  const float* gamma = (const float*)d_in[2];   // [1]
  float* out = (float*)d_out;

  // workspace layout (bf16 intermediates)
  char* ws = (char*)d_ws;
  size_t off = 0;
  bf16_t* xh     = (bf16_t*)(ws + off); off += (size_t)BATCH * CDIM * NDIM * 2; // 128 MB
  bf16_t* xhT    = (bf16_t*)(ws + off); off += (size_t)BATCH * CDIM * NDIM * 2; // 128 MB
  bf16_t* Wh     = (bf16_t*)(ws + off); off += (size_t)CDIM * CDIM * 2;         //   2 MB
  bf16_t* convxT = (bf16_t*)(ws + off); off += (size_t)BATCH * CDIM * NDIM * 2; // 128 MB
  bf16_t* xxT    = (bf16_t*)(ws + off);                                         //  32 MB

  const long strideXN = (long)CDIM * NDIM;   // per-batch x/convxT/out stride
  const long strideCC = (long)CDIM * CDIM;   // per-batch xxT stride

  // one pass over x: produce xh [C x N] and xhT [N x C] in bf16
  cvt_and_transpose<<<dim3(NDIM / 32, CDIM / 32, BATCH), 256, 0, stream>>>(
      x, xh, xhT, CDIM, NDIM);
  // W -> bf16
  {
    long w4 = (long)CDIM * CDIM / 4;
    cvt_f32_bf16<<<(int)((w4 + 255) / 256), 256, 0, stream>>>(W, Wh, w4);
  }

  // Stage 1: convxT[b] (N x C) = xhT[b] (N x C) * Wh^T        [NT, K=C]
  gemm_nt_bf16_wmma<false>
      <<<dim3(CDIM / BN, NDIM / BM, BATCH), 256, 0, stream>>>(
          xhT, strideXN, Wh, 0L, convxT, nullptr, nullptr, strideXN, nullptr,
          CDIM, CDIM, CDIM, CDIM);

  // Stage 2: xxT[b] (C x C) = xh[b] (C x N) * xh[b]^T          [NT, K=N]
  gemm_nt_bf16_wmma<false>
      <<<dim3(CDIM / BN, CDIM / BM, BATCH), 256, 0, stream>>>(
          xh, strideXN, xh, strideXN, xxT, nullptr, nullptr, strideCC, nullptr,
          NDIM, NDIM, NDIM, CDIM);

  // Stage 3: out[b] (C x N) = gamma * (xxT[b] * convxT[b]^T) + x[b]   [NT + epilogue]
  gemm_nt_bf16_wmma<true>
      <<<dim3(NDIM / BN, CDIM / BM, BATCH), 256, 0, stream>>>(
          xxT, strideCC, convxT, strideXN, nullptr, out, x, strideXN, gamma,
          CDIM, CDIM, CDIM, NDIM);
}
